// HRSparseEncoderV2_43361989820742
// MI455X (gfx1250) — matrix operand order repaired and checked by reading, not compiled, not run on hardware
//
#include <hip/hip_runtime.h>
#include <hip/hip_bf16.h>

// ---------------------------------------------------------------------------
// CDNA5 (gfx1250) HR sparse-encoder pipeline.
// Masked dense 3D convs as implicit GEMMs on v_wmma_f32_16x16x32_bf16.
// 2x2 register tiling: one wave computes a 32(voxel) x (16*NT)(channel) tile;
// per chunk {4x b128 A, 2*NT x b128 B, 2*NT WMMAs} -> 2 loads per WMMA.
// Tap-major K loop: neighbor base addresses computed once per tap (OOB taps
// read a zero page -> branch-free hot loop). Channels padded to >=32 so every
// 32-K chunk sits in one tap. No LDS, no div/mod in the hot loop.
// ---------------------------------------------------------------------------

typedef __attribute__((ext_vector_type(16))) __bf16       v16bf;
typedef __attribute__((ext_vector_type(8)))  float        v8f;
typedef __attribute__((ext_vector_type(4)))  unsigned int u32x4;

union FragBF { u32x4 q[2]; unsigned int u[8]; v16bf v; };
union PairBF { __bf16 h[2]; unsigned int u; };

// ------------------------- scatter (stride-32 padded channels) ---------------
__global__ void hr_scatter(const float* __restrict__ vf, const int* __restrict__ coors,
                           __bf16* __restrict__ x, unsigned char* __restrict__ m, int n)
{
    int i = blockIdx.x * blockDim.x + threadIdx.x;
    if (i >= n) return;
    int h = coors[i * 4 + 1];
    int w = coors[i * 4 + 2];
    int d = coors[i * 4 + 3];
    int vox = (h * 128 + w) * 36 + d;
    m[vox] = 1;
    #pragma unroll
    for (int c = 0; c < 4; ++c)
        x[(size_t)vox * 32 + c] = (__bf16)vf[i * 4 + c];
}

// ------------------------- mask dilation ------------------------------------
__global__ void hr_dilate(const unsigned char* __restrict__ mi, unsigned char* __restrict__ mo,
                          int Hi, int Wi, int Di, int Ho, int Wo, int Do,
                          int kh, int kw, int kd, int sh, int sw, int sd,
                          int ph, int pw, int pd)
{
    int v = blockIdx.x * blockDim.x + threadIdx.x;
    int Vout = Ho * Wo * Do;
    if (v >= Vout) return;
    int d = v % Do;
    int t = v / Do;
    int w = t % Wo;
    int h = t / Wo;
    unsigned char r = 0;
    for (int a = 0; a < kh; ++a) {
        int hi = h * sh - ph + a;
        if ((unsigned)hi >= (unsigned)Hi) continue;
        for (int b = 0; b < kw; ++b) {
            int wi = w * sw - pw + b;
            if ((unsigned)wi >= (unsigned)Wi) continue;
            for (int c = 0; c < kd; ++c) {
                int di = d * sd - pd + c;
                if ((unsigned)di >= (unsigned)Di) continue;
                r |= mi[(hi * Wi + wi) * Di + di];
            }
        }
    }
    mo[v] = r ? 1 : 0;
}

// ------------------------- weight pack --------------------------------------
// f32 weights [taps][CIorig][Co] -> bf16-pair u32 tensor laid out for two
// b128 loads per lane: flat = ((((nt*NCH + ch)*2 + khalf)*2 + rq)*16 + n)*4 + rl
// Zero-fills padded channels (c >= CIorig).
__global__ void hr_pack_w(const float* __restrict__ w, unsigned int* __restrict__ pw,
                          int CIorig, int CIpad, int NCH, int Co)
{
    int i = blockIdx.x * blockDim.x + threadIdx.x;
    int total = (Co >> 4) * NCH * 256;
    if (i >= total) return;
    int rl    = i & 3;
    int n     = (i >> 2) & 15;
    int rq    = (i >> 6) & 1;
    int khalf = (i >> 7) & 1;
    int ch    = (i >> 8) % NCH;
    int nt    = i / (NCH * 256);
    int kk  = khalf * 8 + rq * 4 + rl;   // local pair index in chunk
    int kp  = ch * 32 + kk * 2;          // padded k
    int tap = kp / CIpad;
    int c   = kp % CIpad;
    int col = nt * 16 + n;
    PairBF p;
    p.h[0] = (c     < CIorig) ? (__bf16)w[((size_t)tap * CIorig + c    ) * Co + col] : (__bf16)0.f;
    p.h[1] = (c + 1 < CIorig) ? (__bf16)w[((size_t)tap * CIorig + c + 1) * Co + col] : (__bf16)0.f;
    pw[i] = p.u;
}

// ------------------------- implicit-GEMM conv3d (bf16 WMMA) ------------------
// One wave computes a 32(voxel) x (16*NT)(out-channel) tile. Per tap two base
// pointers (zero page if OOB), then CIPAD/32 chunks of
// {4x b128 A, 2*NT x b128 B, 2*NT WMMAs with full A/B fragment reuse}.
template <int CIPAD, int KH, int KW, int KD, int NT>
__global__ __launch_bounds__(256)
void hr_conv3d_wmma(const __bf16* __restrict__ in, __bf16* __restrict__ out,
                    const unsigned int* __restrict__ pw,
                    const float* __restrict__ gam, const float* __restrict__ bet,
                    const unsigned char* __restrict__ mask,
                    const __bf16* __restrict__ zb,
                    int Hi, int Wi, int Di,
                    int Ho, int Wo, int Do, int Co, int CoS,
                    int sh, int sw, int sd,
                    int ph, int pww, int pd)
{
    constexpr int NCH = KH * KW * KD * CIPAD / 32;

    const int lane  = threadIdx.x & 31;
    const int wv    = threadIdx.x >> 5;
    const int mrow  = lane & 15;
    const int khalf = lane >> 4;

    const int Vout   = Ho * Wo * Do;
    const int Mtiles = (Vout + 31) >> 5;           // 32 voxels per wave
    const int Ngrp   = Co / (16 * NT);
    const int tile   = blockIdx.x * 8 + wv;
    if (tile >= Mtiles * Ngrp) return;   // uniform per wave

    const int mt  = tile % Mtiles;
    const int nt0 = (tile / Mtiles) * NT;

    // ---- two M sub-tiles: voxels mt*32+mrow and +16 ----
    const int  vox0 = mt * 32 + mrow;
    const int  vox1 = vox0 + 16;
    const bool vld0 = vox0 < Vout;
    const bool vld1 = vox1 < Vout;
    const int  vv0  = vld0 ? vox0 : 0;
    const int  vv1  = vld1 ? vox1 : 0;
    const int  dA = vv0 % Do, tA = vv0 / Do, wA = tA % Wo, hA = tA / Wo;
    const int  dB = vv1 % Do, tB = vv1 / Do, wB = tB % Wo, hB = tB / Wo;
    const int  h0a = hA * sh - ph,  w0a = wA * sw - pww, d0a = dA * sd - pd;
    const int  h0b = hB * sh - ph,  w0b = wB * sw - pww, d0b = dB * sd - pd;

    const unsigned int* pwp = pw + (size_t)nt0 * NCH * 256 + khalf * 128 + mrow * 4;

    v8f acc00 = {};   // (m-sub 0, n-sub 0)
    v8f acc01 = {};   // (m-sub 0, n-sub 1)
    v8f acc10 = {};   // (m-sub 1, n-sub 0)
    v8f acc11 = {};   // (m-sub 1, n-sub 1)

    for (int ikh = 0; ikh < KH; ++ikh)
    for (int ikw = 0; ikw < KW; ++ikw) {
        #pragma unroll
        for (int ikd = 0; ikd < KD; ++ikd) {
            const int hia = h0a + ikh, wia = w0a + ikw, dia = d0a + ikd;
            const int hib = h0b + ikh, wib = w0b + ikw, dib = d0b + ikd;
            const bool ok0 = vld0 && (unsigned)hia < (unsigned)Hi &&
                             (unsigned)wia < (unsigned)Wi && (unsigned)dia < (unsigned)Di;
            const bool ok1 = vld1 && (unsigned)hib < (unsigned)Hi &&
                             (unsigned)wib < (unsigned)Wi && (unsigned)dib < (unsigned)Di;
            const __bf16* base0 =
                ok0 ? (in + (size_t)((hia * Wi + wia) * Di + dia) * CIPAD) : zb;
            const __bf16* base1 =
                ok1 ? (in + (size_t)((hib * Wi + wib) * Di + dib) * CIPAD) : zb;
            __builtin_prefetch(pwp + 256, 0, 1);  // next chunk of packed weights
            #pragma unroll
            for (int c0 = 0; c0 < CIPAD; c0 += 32) {
                FragBF a0, a1, b0, b1;
                a0.q[0] = *(const u32x4*)(base0 + c0 + 16 * khalf);
                a0.q[1] = *(const u32x4*)(base0 + c0 + 16 * khalf + 8);
                a1.q[0] = *(const u32x4*)(base1 + c0 + 16 * khalf);
                a1.q[1] = *(const u32x4*)(base1 + c0 + 16 * khalf + 8);
                b0.q[0] = *(const u32x4*)(pwp);
                b0.q[1] = *(const u32x4*)(pwp + 64);
                if constexpr (NT == 2) {
                    b1.q[0] = *(const u32x4*)(pwp + NCH * 256);
                    b1.q[1] = *(const u32x4*)(pwp + NCH * 256 + 64);
                }
                pwp += 256;
                acc00 = __builtin_amdgcn_wmma_f32_16x16x32_bf16(
                    false, a0.v, false, b0.v, (short)0, acc00, false, false);
                acc10 = __builtin_amdgcn_wmma_f32_16x16x32_bf16(
                    false, a1.v, false, b0.v, (short)0, acc10, false, false);
                if constexpr (NT == 2) {
                    acc01 = __builtin_amdgcn_wmma_f32_16x16x32_bf16(
                        false, a0.v, false, b1.v, (short)0, acc01, false, false);
                    acc11 = __builtin_amdgcn_wmma_f32_16x16x32_bf16(
                        false, a1.v, false, b1.v, (short)0, acc11, false, false);
                }
            }
        }
    }

    // ---- epilogue: scale/bias, mask, relu, store bf16 (2x2 sub-tiles) ----
    const int   col0 = nt0 * 16 + mrow;
    const float g0   = gam[col0];
    const float bb0  = bet[col0];
    float g1 = 0.f, bb1 = 0.f;
    if constexpr (NT == 2) { g1 = gam[col0 + 16]; bb1 = bet[col0 + 16]; }
    #pragma unroll
    for (int r = 0; r < 8; ++r) {
        const int mr  = r + (khalf ? 8 : 0);
        const int v20 = mt * 32 + mr;        // m-sub 0 voxel
        const int v21 = v20 + 16;            // m-sub 1 voxel
        if (v20 < Vout) {
            const float mk = mask[v20] ? 1.f : 0.f;
            float y = (acc00[r] * g0 + bb0) * mk;
            y = y > 0.f ? y : 0.f;
            out[(size_t)v20 * CoS + col0] = (__bf16)y;
            if constexpr (NT == 2) {
                float z = (acc01[r] * g1 + bb1) * mk;
                z = z > 0.f ? z : 0.f;
                out[(size_t)v20 * CoS + col0 + 16] = (__bf16)z;
            }
        }
        if (v21 < Vout) {
            const float mk = mask[v21] ? 1.f : 0.f;
            float y = (acc10[r] * g0 + bb0) * mk;
            y = y > 0.f ? y : 0.f;
            out[(size_t)v21 * CoS + col0] = (__bf16)y;
            if constexpr (NT == 2) {
                float z = (acc11[r] * g1 + bb1) * mk;
                z = z > 0.f ? z : 0.f;
                out[(size_t)v21 * CoS + col0 + 16] = (__bf16)z;
            }
        }
    }
}

// ------------------------- final transform + sum ----------------------------
__global__ void hr_final(const __bf16* __restrict__ F1, const __bf16* __restrict__ F2,
                         const __bf16* __restrict__ F3, const __bf16* __restrict__ F4,
                         float* __restrict__ out)
{
    int i = blockIdx.x * blockDim.x + threadIdx.x;
    if (i >= 512 * 32 * 32) return;
    int x  = i % 32;
    int y  = (i / 32) % 32;
    int cp = i / (32 * 32);
    float s = 0.f;
    { int c = cp >> 4, d = cp & 15; s += (float)F1[((y * 32 + x) * 16 + d) * 32  + c]; }
    { int c = cp >> 3, d = cp & 7;  s += (float)F2[((y * 32 + x) * 8  + d) * 64  + c]; }
    { int c = cp >> 2, d = cp & 3;  int hs = y >> 1, ws = x >> 1;
      s += (float)F3[((hs * 16 + ws) * 4 + d) * 128 + c]; }
    { int c = cp >> 2, d = cp & 3;  int hs = y >> 1, ws = x >> 1;
      s += (float)F4[((hs * 16 + ws) * 4 + d) * 128 + c]; }
    out[i] = s;
}

// ---------------------------------------------------------------------------
// Host-side orchestration
// ---------------------------------------------------------------------------
namespace {

constexpr int H0 = 128, W0 = 128, D0 = 36; constexpr int V0 = H0 * W0 * D0;
constexpr int H1 = 64,  W1 = 64,  D1 = 18; constexpr int V1 = H1 * W1 * D1;
constexpr int H2 = 32,  W2 = 32,  D2 = 9;  constexpr int V2 = H2 * W2 * D2;
constexpr int H3 = 16,  W3 = 16,  D3 = 5;  constexpr int V3 = H3 * W3 * D3;
constexpr int HF1 = 32, WF1 = 32, DF1 = 16; constexpr int VF1 = HF1 * WF1 * DF1;
constexpr int HF2 = 32, WF2 = 32, DF2 = 8;  constexpr int VF2 = HF2 * WF2 * DF2;
constexpr int HF3 = 16, WF3 = 16, DF3 = 4;  constexpr int VF3 = HF3 * WF3 * DF3;
constexpr int HF4 = 16, WF4 = 16, DF4 = 4;  constexpr int VF4 = HF4 * WF4 * DF4;

struct Bump {
    char*  p;
    size_t off;
    void* take(size_t bytes) {
        size_t a = (off + 255) & ~(size_t)255;
        off = a + bytes;
        return p + a;
    }
};

inline void launch_conv(hipStream_t s, const __bf16* in, __bf16* out,
                        const float* w, const float* g, const float* b,
                        const unsigned char* mask, unsigned int* wpack, const __bf16* zb,
                        int Hi, int Wi, int Di, int CIorig,
                        int Ho, int Wo, int Do, int Co, int CoS,
                        int kh, int kw, int kd,
                        int sh, int sw, int sd,
                        int ph, int pw, int pd)
{
    const int CIpad = CIorig < 32 ? 32 : CIorig;
    const int NCH   = kh * kw * kd * CIpad / 32;
    const int packN = (Co / 16) * NCH * 256;
    hr_pack_w<<<(packN + 255) / 256, 256, 0, s>>>(w, wpack, CIorig, CIpad, NCH, Co);

    const int NT     = (Co >= 32) ? 2 : 1;
    const int Vout   = Ho * Wo * Do;
    const int tiles  = ((Vout + 31) / 32) * (Co / (16 * NT));
    const int blocks = (tiles + 7) / 8;

#define HR_CASE(cp, a_, b_, c_, ntv)                                             \
    if (CIpad == cp && kh == a_ && kw == b_ && kd == c_ && NT == ntv) {          \
        hr_conv3d_wmma<cp, a_, b_, c_, ntv><<<blocks, 256, 0, s>>>(              \
            in, out, wpack, g, b, mask, zb, Hi, Wi, Di, Ho, Wo, Do, Co, CoS,     \
            sh, sw, sd, ph, pw, pd);                                             \
        return;                                                                  \
    }
    HR_CASE(32,  3, 3, 3, 1)   // enc block 0 (Co=16)
    HR_CASE(32,  3, 3, 3, 2)
    HR_CASE(64,  3, 3, 3, 2)
    HR_CASE(128, 3, 3, 3, 2)
    HR_CASE(32,  1, 1, 6, 2)
    HR_CASE(64,  1, 1, 4, 2)
    HR_CASE(128, 1, 1, 3, 2)
    HR_CASE(128, 1, 1, 2, 2)
#undef HR_CASE
}

inline void launch_dilate(hipStream_t s, const unsigned char* mi, unsigned char* mo,
                          int Hi, int Wi, int Di, int Ho, int Wo, int Do,
                          int kh, int kw, int kd, int sh, int sw, int sd,
                          int ph, int pw, int pd)
{
    int Vout = Ho * Wo * Do;
    hr_dilate<<<(Vout + 255) / 256, 256, 0, s>>>(mi, mo, Hi, Wi, Di, Ho, Wo, Do,
                                                 kh, kw, kd, sh, sw, sd, ph, pw, pd);
}

} // namespace

extern "C" void kernel_launch(void* const* d_in, const int* in_sizes, int n_in,
                              void* d_out, int out_size, void* d_ws, size_t ws_size,
                              hipStream_t stream)
{
    // ---- inputs (setup_inputs dict order, leaves flattened depth-first) ----
    const float* vf    = (const float*)d_in[0];
    const int*   coors = (const int*)d_in[70];
    auto P = [&](int i) { return (const float*)d_in[i]; };
    // enc_params: (w,g,b) x 11 at 1..33 ; br_params at 34..57 ; fuse at 58..69

    // ---- workspace layout ----
    Bump ws{(char*)d_ws, 0};
    __bf16* x_in0 = (__bf16*)ws.take((size_t)V0 * 32 * 2);   // CI=4 padded to 32
    __bf16* xA0   = (__bf16*)ws.take((size_t)V0 * 32 * 2);   // CI=16 padded to 32
    __bf16* xB0   = (__bf16*)ws.take((size_t)V0 * 32 * 2);   // feat0
    __bf16* xC0   = (__bf16*)ws.take((size_t)V0 * 32 * 2);
    unsigned char* m0  = (unsigned char*)ws.take(V0);
    unsigned char* m0b = (unsigned char*)ws.take(V0);
    __bf16* y1in = (__bf16*)ws.take((size_t)V1 * 64 * 2);
    __bf16* y1a  = (__bf16*)ws.take((size_t)V1 * 64 * 2);
    __bf16* y1b  = (__bf16*)ws.take((size_t)V1 * 64 * 2);    // feat1
    __bf16* y1c  = (__bf16*)ws.take((size_t)V1 * 64 * 2);
    unsigned char* m1  = (unsigned char*)ws.take(V1);
    unsigned char* m1b = (unsigned char*)ws.take(V1);
    __bf16* y2in = (__bf16*)ws.take((size_t)V2 * 128 * 2);
    __bf16* y2a  = (__bf16*)ws.take((size_t)V2 * 128 * 2);
    __bf16* y2b  = (__bf16*)ws.take((size_t)V2 * 128 * 2);   // feat2
    __bf16* y2c  = (__bf16*)ws.take((size_t)V2 * 128 * 2);
    __bf16* y2d  = (__bf16*)ws.take((size_t)V2 * 128 * 2);
    unsigned char* m2  = (unsigned char*)ws.take(V2);
    unsigned char* m2b = (unsigned char*)ws.take(V2);
    unsigned char* m2c = (unsigned char*)ws.take(V2);
    __bf16* y3in = (__bf16*)ws.take((size_t)V3 * 128 * 2);
    __bf16* y3a  = (__bf16*)ws.take((size_t)V3 * 128 * 2);
    __bf16* y3b  = (__bf16*)ws.take((size_t)V3 * 128 * 2);   // feat3
    __bf16* y3c  = (__bf16*)ws.take((size_t)V3 * 128 * 2);
    __bf16* y3d  = (__bf16*)ws.take((size_t)V3 * 128 * 2);
    unsigned char* m3  = (unsigned char*)ws.take(V3);
    unsigned char* m3b = (unsigned char*)ws.take(V3);
    unsigned char* m3c = (unsigned char*)ws.take(V3);
    unsigned char* m3d = (unsigned char*)ws.take(V3);
    unsigned char* m3e = (unsigned char*)ws.take(V3);
    unsigned char* mF1 = (unsigned char*)ws.take(VF1);
    unsigned char* mF2 = (unsigned char*)ws.take(VF2);
    unsigned char* mF3 = (unsigned char*)ws.take(VF3);
    unsigned char* mF4 = (unsigned char*)ws.take(VF4);
    __bf16* F1 = (__bf16*)ws.take((size_t)VF1 * 32  * 2);
    __bf16* F2 = (__bf16*)ws.take((size_t)VF2 * 64  * 2);
    __bf16* F3 = (__bf16*)ws.take((size_t)VF3 * 128 * 2);
    __bf16* F4 = (__bf16*)ws.take((size_t)VF4 * 128 * 2);
    unsigned int* wpack = (unsigned int*)ws.take((size_t)8 * 108 * 256 * 4); // max pack: 884 KB
    __bf16* zb = (__bf16*)ws.take(1024);                     // zero page for OOB taps
    (void)ws_size; (void)n_in; (void)out_size;

    // ---- zero-init: scatter target, padded-channel buffer, zero page ----
    hipMemsetAsync(x_in0, 0, (size_t)V0 * 32 * 2, stream);
    hipMemsetAsync(xA0,   0, (size_t)V0 * 32 * 2, stream);
    hipMemsetAsync(m0,    0, V0, stream);
    hipMemsetAsync(zb,    0, 1024, stream);
    int nvox = in_sizes[0] / 4;
    hr_scatter<<<(nvox + 255) / 256, 256, 0, stream>>>(vf, coors, x_in0, m0, nvox);

    // ---- stage 0 (128x128x36) ----
    launch_conv(stream, x_in0, xA0, P(1),  P(2),  P(3),  m0, wpack, zb, H0,W0,D0,4,  H0,W0,D0,16,32, 3,3,3, 1,1,1, 1,1,1);
    launch_conv(stream, xA0,   xB0, P(4),  P(5),  P(6),  m0, wpack, zb, H0,W0,D0,16, H0,W0,D0,32,32, 3,3,3, 1,1,1, 1,1,1);
    // branch 0 (1 block) + fuse 0
    launch_dilate(stream, m0,  m0b, H0,W0,D0, H0,W0,D0, 3,3,3, 1,1,1, 1,1,1);
    launch_conv(stream, xB0, xC0, P(34), P(35), P(36), m0b, wpack, zb, H0,W0,D0,32, H0,W0,D0,32,32, 3,3,3, 1,1,1, 1,1,1);
    launch_dilate(stream, m0b, mF1, H0,W0,D0, HF1,WF1,DF1, 1,1,6, 4,4,2, 0,0,0);
    launch_conv(stream, xC0, F1, P(58), P(59), P(60), mF1, wpack, zb, H0,W0,D0,32, HF1,WF1,DF1,32,32, 1,1,6, 4,4,2, 0,0,0);

    // ---- stage 1 (64x64x18) ----
    launch_dilate(stream, m0, m1, H0,W0,D0, H1,W1,D1, 3,3,3, 2,2,2, 1,1,1);
    launch_conv(stream, xB0,  y1in, P(7),  P(8),  P(9),  m1, wpack, zb, H0,W0,D0,32, H1,W1,D1,64,64, 3,3,3, 2,2,2, 1,1,1);
    launch_conv(stream, y1in, y1a,  P(10), P(11), P(12), m1, wpack, zb, H1,W1,D1,64, H1,W1,D1,64,64, 3,3,3, 1,1,1, 1,1,1);
    launch_conv(stream, y1a,  y1b,  P(13), P(14), P(15), m1, wpack, zb, H1,W1,D1,64, H1,W1,D1,64,64, 3,3,3, 1,1,1, 1,1,1);
    // branch 1 + fuse 1
    launch_dilate(stream, m1, m1b, H1,W1,D1, H1,W1,D1, 3,3,3, 1,1,1, 1,1,1);
    launch_conv(stream, y1b, y1c, P(37), P(38), P(39), m1b, wpack, zb, H1,W1,D1,64, H1,W1,D1,64,64, 3,3,3, 1,1,1, 1,1,1);
    launch_dilate(stream, m1b, mF2, H1,W1,D1, HF2,WF2,DF2, 1,1,4, 2,2,2, 0,0,0);
    launch_conv(stream, y1c, F2, P(61), P(62), P(63), mF2, wpack, zb, H1,W1,D1,64, HF2,WF2,DF2,64,64, 1,1,4, 2,2,2, 0,0,0);

    // ---- stage 2 (32x32x9) ----
    launch_dilate(stream, m1, m2, H1,W1,D1, H2,W2,D2, 3,3,3, 2,2,2, 1,1,1);
    launch_conv(stream, y1b,  y2in, P(16), P(17), P(18), m2, wpack, zb, H1,W1,D1,64,  H2,W2,D2,128,128, 3,3,3, 2,2,2, 1,1,1);
    launch_conv(stream, y2in, y2a,  P(19), P(20), P(21), m2, wpack, zb, H2,W2,D2,128, H2,W2,D2,128,128, 3,3,3, 1,1,1, 1,1,1);
    launch_conv(stream, y2a,  y2b,  P(22), P(23), P(24), m2, wpack, zb, H2,W2,D2,128, H2,W2,D2,128,128, 3,3,3, 1,1,1, 1,1,1);
    // branch 2 (2 blocks) + fuse 2
    launch_dilate(stream, m2,  m2b, H2,W2,D2, H2,W2,D2, 3,3,3, 1,1,1, 1,1,1);
    launch_conv(stream, y2b, y2c, P(40), P(41), P(42), m2b, wpack, zb, H2,W2,D2,128, H2,W2,D2,128,128, 3,3,3, 1,1,1, 1,1,1);
    launch_dilate(stream, m2b, m2c, H2,W2,D2, H2,W2,D2, 3,3,3, 1,1,1, 1,1,1);
    launch_conv(stream, y2c, y2d, P(43), P(44), P(45), m2c, wpack, zb, H2,W2,D2,128, H2,W2,D2,128,128, 3,3,3, 1,1,1, 1,1,1);
    launch_dilate(stream, m2c, mF3, H2,W2,D2, HF3,WF3,DF3, 1,1,3, 2,2,2, 0,0,0);
    launch_conv(stream, y2d, F3, P(64), P(65), P(66), mF3, wpack, zb, H2,W2,D2,128, HF3,WF3,DF3,128,128, 1,1,3, 2,2,2, 0,0,0);

    // ---- stage 3 (16x16x5) ----
    launch_dilate(stream, m2, m3, H2,W2,D2, H3,W3,D3, 3,3,3, 2,2,2, 1,1,1);
    launch_conv(stream, y2b,  y3in, P(25), P(26), P(27), m3, wpack, zb, H2,W2,D2,128, H3,W3,D3,128,128, 3,3,3, 2,2,2, 1,1,1);
    launch_conv(stream, y3in, y3a,  P(28), P(29), P(30), m3, wpack, zb, H3,W3,D3,128, H3,W3,D3,128,128, 3,3,3, 1,1,1, 1,1,1);
    launch_conv(stream, y3a,  y3b,  P(31), P(32), P(33), m3, wpack, zb, H3,W3,D3,128, H3,W3,D3,128,128, 3,3,3, 1,1,1, 1,1,1);
    // branch 3 (4 blocks) + fuse 3
    launch_dilate(stream, m3,  m3b, H3,W3,D3, H3,W3,D3, 3,3,3, 1,1,1, 1,1,1);
    launch_conv(stream, y3b, y3c, P(46), P(47), P(48), m3b, wpack, zb, H3,W3,D3,128, H3,W3,D3,128,128, 3,3,3, 1,1,1, 1,1,1);
    launch_dilate(stream, m3b, m3c, H3,W3,D3, H3,W3,D3, 3,3,3, 1,1,1, 1,1,1);
    launch_conv(stream, y3c, y3d, P(49), P(50), P(51), m3c, wpack, zb, H3,W3,D3,128, H3,W3,D3,128,128, 3,3,3, 1,1,1, 1,1,1);
    launch_dilate(stream, m3c, m3d, H3,W3,D3, H3,W3,D3, 3,3,3, 1,1,1, 1,1,1);
    launch_conv(stream, y3d, y3c, P(52), P(53), P(54), m3d, wpack, zb, H3,W3,D3,128, H3,W3,D3,128,128, 3,3,3, 1,1,1, 1,1,1);
    launch_dilate(stream, m3d, m3e, H3,W3,D3, H3,W3,D3, 3,3,3, 1,1,1, 1,1,1);
    launch_conv(stream, y3c, y3d, P(55), P(56), P(57), m3e, wpack, zb, H3,W3,D3,128, H3,W3,D3,128,128, 3,3,3, 1,1,1, 1,1,1);
    launch_dilate(stream, m3e, mF4, H3,W3,D3, HF4,WF4,DF4, 1,1,2, 1,1,1, 0,0,0);
    launch_conv(stream, y3d, F4, P(67), P(68), P(69), mF4, wpack, zb, H3,W3,D3,128, HF4,WF4,DF4,128,128, 1,1,2, 1,1,1, 0,0,0);

    // ---- final transform + sum -> d_out (f32, 512*32*32) ----
    hr_final<<<(512 * 32 * 32 + 255) / 256, 256, 0, stream>>>(F1, F2, F3, F4, (float*)d_out);
}
